// RNNUsingSimpleRNNCell_35811437314455
// MI455X (gfx1250) — compile-verified, matrix operand order checked by reading, not run on hardware
//
#include <hip/hip_runtime.h>
#include <math.h>

// ---------------------------------------------------------------------------
// Problem geometry (from the reference)
// ---------------------------------------------------------------------------
#define BATCH 128
#define SEQ   80
#define EMBD  100
#define UNITS 2048
#define KP0   128                 // EMB padded to a multiple of 32 (WMMA K)
#define BN    32                  // output columns per workgroup
#define NBLK  (UNITS / BN)        // 64 workgroups (persistent)
#define TPB   256                 // 8 wave32 per workgroup

// LDS: per-WG strips of W1^T and U1^T (32 rows x 2048 bf16 each), row-padded
// by 16B so that the 16 lanes of a B-fragment ds_load_b128 hit disjoint
// 4-bank groups (conflict free on 64 banks x 4B).
#define LDS_ROW_STRIDE (UNITS * 2 + 16)        // 4112 bytes, 16B aligned
#define LDS_U_OFF      (BN * LDS_ROW_STRIDE)   // 131584
#define LDS_BYTES      (2 * BN * LDS_ROW_STRIDE) // 263168 < 320KB

typedef __attribute__((ext_vector_type(16))) __bf16       v16bf;
typedef __attribute__((ext_vector_type(8)))  float        v8f;
typedef __attribute__((ext_vector_type(4)))  unsigned int u32x4;

union FragBF { v16bf v; u32x4 u[2]; };

__device__ __forceinline__ unsigned short f2bf(float f) {   // RNE f32 -> bf16
  unsigned x = __float_as_uint(f);
  unsigned r = ((x >> 16) & 1u) + 0x7FFFu;
  return (unsigned short)((x + r) >> 16);
}
__device__ __forceinline__ float bf2f(unsigned short h) {
  return __uint_as_float(((unsigned)h) << 16);
}

__device__ __forceinline__ v8f wmma_bf16(const FragBF& a, const FragBF& b, v8f c) {
  // D(f32 16x16) = A(16x32 bf16) * B(32x16 bf16) + C
  return __builtin_amdgcn_wmma_f32_16x16x32_bf16(
      false, a.v, false, b.v, (short)0, c, false, false);
}

// GEMM inner loop, B fragments from global memory (per-lane bases, row-major
// over K, stride already folded into the base pointers).
__device__ __forceinline__ void gemm_glb(v8f& c0, v8f& c1,
    const unsigned short* __restrict__ A,
    const unsigned short* __restrict__ B0,
    const unsigned short* __restrict__ B1, int K) {
#pragma unroll 4
  for (int k = 0; k < K; k += 32) {
    FragBF a, b0, b1;
    a.u[0]  = *(const u32x4*)(A  + k);
    a.u[1]  = *(const u32x4*)(A  + k + 16);
    b0.u[0] = *(const u32x4*)(B0 + k);
    b0.u[1] = *(const u32x4*)(B0 + k + 16);
    b1.u[0] = *(const u32x4*)(B1 + k);
    b1.u[1] = *(const u32x4*)(B1 + k + 16);
    c0 = wmma_bf16(a, b0, c0);
    c1 = wmma_bf16(a, b1, c1);
  }
}

// GEMM inner loop, B fragments from LDS (byte pointers, padded rows).
__device__ __forceinline__ void gemm_ldsB(v8f& c0, v8f& c1,
    const unsigned short* __restrict__ A,
    const char* B0, const char* B1, int K) {
#pragma unroll 4
  for (int k = 0; k < K; k += 32) {
    FragBF a, b0, b1;
    a.u[0]  = *(const u32x4*)(A + k);
    a.u[1]  = *(const u32x4*)(A + k + 16);
    b0.u[0] = *(const u32x4*)(B0 + 2 * k);
    b0.u[1] = *(const u32x4*)(B0 + 2 * k + 32);
    b1.u[0] = *(const u32x4*)(B1 + 2 * k);
    b1.u[1] = *(const u32x4*)(B1 + 2 * k + 32);
    c0 = wmma_bf16(a, b0, c0);
    c1 = wmma_bf16(a, b1, c1);
  }
}

// bias + tanh + bf16 store of two 16x16 C tiles (ISA C-layout: VGPR r holds
// M = r (lanes 0-15) / r+8 (lanes 16-31), N = lane&15 within the tile).
__device__ __forceinline__ void finish_store(const v8f& c0, const v8f& c1,
    const float* __restrict__ bias, unsigned short* __restrict__ Hn,
    int tnb, int tm, int lane) {
  const int n0 = tnb + (lane & 15);
  const int n1 = n0 + 16;
  const int mb = tm * 16 + (lane >> 4) * 8;
  const float bv0 = bias[n0];
  const float bv1 = bias[n1];
#pragma unroll
  for (int r = 0; r < 8; ++r) {
    Hn[(size_t)(mb + r) * UNITS + n0] = f2bf(tanhf(c0[r] + bv0));
    Hn[(size_t)(mb + r) * UNITS + n1] = f2bf(tanhf(c1[r] + bv1));
  }
}

// Device-wide barrier (monotonic counter in workspace; 64 co-resident WGs).
__device__ __forceinline__ void grid_barrier(unsigned* __restrict__ bar,
                                             unsigned* phase) {
  __threadfence();
  __syncthreads();
  if (threadIdx.x == 0) {
    const unsigned target = (*phase) * (unsigned)NBLK + (unsigned)NBLK;
    __hip_atomic_fetch_add(bar, 1u, __ATOMIC_RELEASE, __HIP_MEMORY_SCOPE_AGENT);
    while (__hip_atomic_load(bar, __ATOMIC_ACQUIRE, __HIP_MEMORY_SCOPE_AGENT) < target) {
      __builtin_amdgcn_s_sleep(2);
    }
  }
  __syncthreads();
  __threadfence();
  ++(*phase);
}

// ---------------------------------------------------------------------------
// Prep kernels
// ---------------------------------------------------------------------------
__global__ void k_zero_u32(unsigned* __restrict__ p, unsigned n) {
  unsigned i = blockIdx.x * blockDim.x + threadIdx.x;
  if (i < n) p[i] = 0u;
}

// W[K][N] f32  ->  Bt[N][K] bf16  (Kpad == K, coalesced reads)
__global__ void k_transpose_fast(const float* __restrict__ W,
                                 unsigned short* __restrict__ Bt,
                                 int K, int N) {
  size_t i = (size_t)blockIdx.x * blockDim.x + threadIdx.x;
  if (i >= (size_t)K * N) return;
  int k = (int)(i / N);
  int n = (int)(i % N);
  Bt[(size_t)n * K + k] = f2bf(W[i]);
}

// W[K][N] f32 -> Bt[N][Kpad] bf16 with zero K-padding
__global__ void k_transpose_pad(const float* __restrict__ W,
                                unsigned short* __restrict__ Bt,
                                int K, int N, int Kpad) {
  size_t i = (size_t)blockIdx.x * blockDim.x + threadIdx.x;
  if (i >= (size_t)N * Kpad) return;
  int n = (int)(i / Kpad);
  int k = (int)(i % Kpad);
  float v = (k < K) ? W[(size_t)k * N + n] : 0.0f;
  Bt[i] = f2bf(v);
}

// embedding gather: x[t][b][kp] bf16, kp padded 100 -> 128 with zeros
__global__ void k_embed(const int* __restrict__ tokens,
                        const float* __restrict__ emb,
                        unsigned short* __restrict__ x) {
  size_t i = (size_t)blockIdx.x * blockDim.x + threadIdx.x;
  if (i >= (size_t)SEQ * BATCH * KP0) return;
  int kp = (int)(i & (KP0 - 1));
  int bt = (int)(i >> 7);
  int b  = bt & (BATCH - 1);
  int t  = bt >> 7;
  int tok = tokens[b * SEQ + t];
  float v = (kp < EMBD) ? emb[(size_t)tok * EMBD + kp] : 0.0f;
  x[i] = f2bf(v);
}

// ---------------------------------------------------------------------------
// Persistent RNN kernel: 64 WGs x 256 threads, weights strip resident in LDS
// ---------------------------------------------------------------------------
__global__ void __launch_bounds__(TPB, 1)
k_rnn_persistent(const unsigned short* __restrict__ xseq,
                 const unsigned short* __restrict__ W0t,
                 const unsigned short* __restrict__ U0t,
                 const unsigned short* __restrict__ W1t,
                 const unsigned short* __restrict__ U1t,
                 const float* __restrict__ b0,
                 const float* __restrict__ b1,
                 unsigned short* __restrict__ hbuf,   // [layer][parity][B][U]
                 unsigned* __restrict__ bar) {
  extern __shared__ char smem[];
  const int tid  = threadIdx.x;
  const int lane = tid & 31;
  const int tm   = tid >> 5;              // wave id -> M-tile 0..7
  const int tnb  = blockIdx.x * BN;       // this WG's output-column base

  // Stage this WG's 32-column strips of W1^T and U1^T into LDS once;
  // reused for layers 1..3 across all 80 timesteps (480 GEMM passes).
  for (int r = 0; r < BN; ++r) {
    ((u32x4*)(smem + (size_t)r * LDS_ROW_STRIDE))[tid] =
        ((const u32x4*)(W1t + (size_t)(tnb + r) * UNITS))[tid];
    ((u32x4*)(smem + LDS_U_OFF + (size_t)r * LDS_ROW_STRIDE))[tid] =
        ((const u32x4*)(U1t + (size_t)(tnb + r) * UNITS))[tid];
  }
  __syncthreads();

  const int row16 = lane & 15;
  const int koff  = (lane >> 4) * 8;      // ISA 16-bit A/B fragment lane split

  // Per-lane LDS B-fragment bases for the two 16-wide N tiles.
  const char* lw0 = smem + (size_t)row16        * LDS_ROW_STRIDE + 2 * koff;
  const char* lw1 = smem + (size_t)(16 + row16) * LDS_ROW_STRIDE + 2 * koff;
  const char* lu0 = smem + LDS_U_OFF + (size_t)row16        * LDS_ROW_STRIDE + 2 * koff;
  const char* lu1 = smem + LDS_U_OFF + (size_t)(16 + row16) * LDS_ROW_STRIDE + 2 * koff;

  // Per-lane global B bases for layer 0 (streams from L2 each step).
  const unsigned short* gw0_0 = W0t + (size_t)(tnb + row16)      * KP0   + koff;
  const unsigned short* gw0_1 = W0t + (size_t)(tnb + 16 + row16) * KP0   + koff;
  const unsigned short* gu0_0 = U0t + (size_t)(tnb + row16)      * UNITS + koff;
  const unsigned short* gu0_1 = U0t + (size_t)(tnb + 16 + row16) * UNITS + koff;

  const size_t HSZ  = (size_t)BATCH * UNITS;
  const size_t arow = (size_t)(tm * 16 + row16);

  unsigned phase = 0;
  for (int t = 0; t < SEQ; ++t) {
    const int cur = t & 1, nxt = cur ^ 1;

    // ---- layer 0: h0' = tanh(x_t @ W0 + h0 @ U0 + b0)
    {
      v8f c0 = {}, c1 = {};
      const unsigned short* ax =
          xseq + (size_t)t * BATCH * KP0 + arow * KP0 + koff;
      gemm_glb(c0, c1, ax, gw0_0, gw0_1, KP0);
      const unsigned short* ah = hbuf + (0 * 2 + cur) * HSZ + arow * UNITS + koff;
      gemm_glb(c0, c1, ah, gu0_0, gu0_1, UNITS);
      finish_store(c0, c1, b0, hbuf + (0 * 2 + nxt) * HSZ, tnb, tm, lane);
    }
    grid_barrier(bar, &phase);

    // ---- layers 1..3 (shared W1/U1/b1), B strips resident in LDS
#pragma unroll 1
    for (int L = 1; L <= 3; ++L) {
      v8f c0 = {}, c1 = {};
      const unsigned short* ain =
          hbuf + ((size_t)(L - 1) * 2 + nxt) * HSZ + arow * UNITS + koff;
      gemm_ldsB(c0, c1, ain, lw0, lw1, UNITS);
      const unsigned short* asf =
          hbuf + ((size_t)L * 2 + cur) * HSZ + arow * UNITS + koff;
      gemm_ldsB(c0, c1, asf, lu0, lu1, UNITS);
      finish_store(c0, c1, b1, hbuf + ((size_t)L * 2 + nxt) * HSZ, tnb, tm, lane);
      grid_barrier(bar, &phase);
    }
  }
}

// ---------------------------------------------------------------------------
// Output head: sigmoid(h3 @ Wo + bo), one block per batch row
// ---------------------------------------------------------------------------
__global__ void k_head(const unsigned short* __restrict__ h3,
                       const float* __restrict__ Wo,
                       const float* __restrict__ bo,
                       float* __restrict__ out) {
  __shared__ float red[TPB];
  const int b = blockIdx.x;
  float s = 0.0f;
  for (int k = threadIdx.x; k < UNITS; k += TPB)
    s += bf2f(h3[(size_t)b * UNITS + k]) * Wo[k];
  red[threadIdx.x] = s;
  __syncthreads();
  for (int off = TPB / 2; off > 0; off >>= 1) {
    if (threadIdx.x < off) red[threadIdx.x] += red[threadIdx.x + off];
    __syncthreads();
  }
  if (threadIdx.x == 0) out[b] = 1.0f / (1.0f + expf(-(red[0] + bo[0])));
}

// ---------------------------------------------------------------------------
extern "C" void kernel_launch(void* const* d_in, const int* in_sizes, int n_in,
                              void* d_out, int out_size, void* d_ws, size_t ws_size,
                              hipStream_t stream) {
  (void)in_sizes; (void)n_in; (void)out_size; (void)ws_size;
  const int*   tokens = (const int*)  d_in[0];
  const float* emb    = (const float*)d_in[1];
  const float* W0     = (const float*)d_in[2];
  const float* U0     = (const float*)d_in[3];
  const float* b0     = (const float*)d_in[4];
  const float* W1     = (const float*)d_in[5];
  const float* U1     = (const float*)d_in[6];
  const float* b1     = (const float*)d_in[7];
  const float* Wo     = (const float*)d_in[8];
  const float* bo     = (const float*)d_in[9];
  float* out = (float*)d_out;

  char* ws = (char*)d_ws;
  size_t off = 0;
  unsigned*       bar  = (unsigned*)(ws + off);       off += 256;
  unsigned short* W0t  = (unsigned short*)(ws + off); off += (size_t)UNITS * KP0   * 2;
  unsigned short* U0t  = (unsigned short*)(ws + off); off += (size_t)UNITS * UNITS * 2;
  unsigned short* W1t  = (unsigned short*)(ws + off); off += (size_t)UNITS * UNITS * 2;
  unsigned short* U1t  = (unsigned short*)(ws + off); off += (size_t)UNITS * UNITS * 2;
  unsigned short* xseq = (unsigned short*)(ws + off); off += (size_t)SEQ * BATCH * KP0 * 2;
  unsigned short* hbuf = (unsigned short*)(ws + off); off += 8ull * BATCH * UNITS * 2;

  // ws is poisoned 0xAA and never re-initialized between replays: reset
  // the barrier counter and the h state every call (deterministic).
  k_zero_u32<<<1, 64, 0, stream>>>(bar, 64);
  {
    const unsigned n = 8u * BATCH * UNITS / 2;  // u32 count of hbuf
    k_zero_u32<<<(n + TPB - 1) / TPB, TPB, 0, stream>>>((unsigned*)hbuf, n);
  }

  // fp32 -> bf16 weight transposes (N-major so B fragments are row-major in K)
  k_transpose_pad<<<((size_t)UNITS * KP0 + TPB - 1) / TPB, TPB, 0, stream>>>(
      W0, W0t, EMBD, UNITS, KP0);
  {
    const size_t n = (size_t)UNITS * UNITS;
    const unsigned g = (unsigned)((n + TPB - 1) / TPB);
    k_transpose_fast<<<g, TPB, 0, stream>>>(U0, U0t, UNITS, UNITS);
    k_transpose_fast<<<g, TPB, 0, stream>>>(W1, W1t, UNITS, UNITS);
    k_transpose_fast<<<g, TPB, 0, stream>>>(U1, U1t, UNITS, UNITS);
  }
  {
    const size_t n = (size_t)SEQ * BATCH * KP0;
    k_embed<<<(unsigned)((n + TPB - 1) / TPB), TPB, 0, stream>>>(tokens, emb, xseq);
  }

  // Persistent recurrent kernel: all 80 steps x 4 layers, grid barriers between
  // layers, W1/U1 strips resident in 263KB of LDS per WGP.
  k_rnn_persistent<<<NBLK, TPB, LDS_BYTES, stream>>>(
      xseq, W0t, U0t, W1t, U1t, b0, b1, hbuf, bar);

  // final h3 parity: t=79 writes nxt=0  ->  hbuf slot (layer 3, parity 0)
  k_head<<<BATCH, TPB, 0, stream>>>(hbuf + 6ull * BATCH * UNITS, Wo, bo, out);
}
